// MixtralDecoderLayer_78434692759606
// MI455X (gfx1250) — matrix operand order repaired and checked by reading, not compiled, not run on hardware
//
#include <hip/hip_runtime.h>
#include <cstdint>
#include <cstddef>

// ---------------------------------------------------------------------------
// Types for CDNA5 WMMA (wave32)
// ---------------------------------------------------------------------------
typedef __attribute__((ext_vector_type(16))) __bf16 v16bf;
typedef __attribute__((ext_vector_type(8)))  __bf16 v8bf;
typedef __attribute__((ext_vector_type(8)))  float  v8f;

// Problem constants (from reference)
constexpr int S_  = 2048;
constexpr int H_  = 2048;
constexpr int NH_ = 16;
constexpr int NKV_= 4;
constexpr int HD_ = 128;
constexpr int F_  = 4096;
constexpr int E_  = 8;
constexpr int T_  = S_;            // B == 1

constexpr int TILE_M = 128;
constexpr int TILE_N = 128;
constexpr int TILE_K = 32;
constexpr int LDSS   = 40;         // padded K-stride in bf16 elements (80B, 16B-aligned)

// ---------------------------------------------------------------------------
// CDNA5 async global->LDS copy (ASYNCcnt-tracked, no VGPR staging).
// Copies 32B per lane: two b128 async ops at +0 and +16 (INST_OFFSET applies
// to both the LDS and global side per ISA 08 §4.4).
// ---------------------------------------------------------------------------
__device__ __forceinline__ void async_copy_32B(unsigned lds_addr, const void* gptr) {
  unsigned long long ga = (unsigned long long)gptr;
  asm volatile("global_load_async_to_lds_b128 %0, %1, off\n\t"
               "global_load_async_to_lds_b128 %0, %1, off offset:16"
               :: "v"(lds_addr), "v"(ga) : "memory");
}
__device__ __forceinline__ void wait_async0() {
  asm volatile("s_wait_asynccnt 0x0" ::: "memory");
}

// ---------------------------------------------------------------------------
// Fragment loads from LDS per ISA 7.12.2 layouts
// A 16x32 bf16: lane L -> row L%16; elems 0..7 = K (L>=16?8:0)+0..7,
//               elems 8..15 = K 16+(L>=16?8:0)+0..7
// B 32x16 bf16 stored transposed in LDS as [N][K]:
//               lane L -> col L%16; elems 0..7 = K (L>=16?8:0)+0..7,
//               elems 8..15 = K 16+(L>=16?8:0)+0..7
// ---------------------------------------------------------------------------
__device__ __forceinline__ v16bf load_frag_a(const __bf16* tile, int lane) {
  int row  = lane & 15;
  int koff = (lane >> 4) << 3;                 // 0 or 8
  const v8bf* p0 = (const v8bf*)(tile + row * LDSS + koff);
  const v8bf* p1 = (const v8bf*)(tile + row * LDSS + koff + 16);
  v8bf lo = *p0, hi = *p1;
  return __builtin_shufflevector(lo, hi, 0,1,2,3,4,5,6,7,8,9,10,11,12,13,14,15);
}

__device__ __forceinline__ v16bf load_frag_b(const __bf16* tile, int lane) {
  int n    = lane & 15;
  int koff = (lane >> 4) << 3;                 // 0 or 8
  const v8bf* p0 = (const v8bf*)(tile + n * LDSS + koff);
  const v8bf* p1 = (const v8bf*)(tile + n * LDSS + koff + 16);
  v8bf lo = *p0, hi = *p1;
  return __builtin_shufflevector(lo, hi, 0,1,2,3,4,5,6,7,8,9,10,11,12,13,14,15);
}

// ---------------------------------------------------------------------------
// Tiled WMMA GEMM with double-buffered LDS pipeline.
//   C[M,N] = A[M,K](bf16,row-major) * B
//   NT == false : B fp32 row-major [K,N] (weights; VGPR-staged, bf16-converted,
//                 software-pipelined so the global loads hide under WMMA)
//   NT == true  : B bf16 row-major [N,K] (async-copied like A)
// 256 threads = 8 waves; block tile 128x128; wave tile 32x64 (2x4 fragments).
// ---------------------------------------------------------------------------
template <bool NT>
__global__ __launch_bounds__(256)
void gemm_wmma_kernel(const __bf16* __restrict__ A, int lda,
                      const void* __restrict__ Bv, int ldb,
                      float* __restrict__ C, int ldc,
                      int M, int N, int K)
{
  __shared__ __attribute__((aligned(16))) __bf16 As[2][TILE_M * LDSS];
  __shared__ __attribute__((aligned(16))) __bf16 Bs[2][TILE_N * LDSS];

  const int tid  = threadIdx.x;
  const int lane = tid & 31;
  const int wave = tid >> 5;
  const int wm   = wave & 3;                   // 4 waves along M (32 rows each)
  const int wn   = wave >> 2;                  // 2 waves along N (64 cols each)
  const int bm   = blockIdx.y * TILE_M;
  const int bn   = blockIdx.x * TILE_N;

  v8f acc[2][4] = {};

  // Tile-loader mappings (256 thr x 32B = one 128x32 bf16 tile)
  const int ar = tid >> 1;                     // row 0..127
  const int ah = (tid & 1) << 4;               // col 0 or 16
  // NN fp32 B mapping: 256 thr x 16 f32 = 32x128
  const int bk = tid >> 3;                     // k-row 0..31
  const int bs = (tid & 7) << 4;               // col segment of 16

  const unsigned ldsA[2] = {
      (unsigned)(size_t)&As[0][ar * LDSS + ah],
      (unsigned)(size_t)&As[1][ar * LDSS + ah] };
  const unsigned ldsB[2] = {
      (unsigned)(size_t)&Bs[0][ar * LDSS + ah],
      (unsigned)(size_t)&Bs[1][ar * LDSS + ah] };

  const __bf16* gA   = A + (size_t)(bm + ar) * lda + ah;
  const __bf16* gBnt = (const __bf16*)Bv + (size_t)(bn + ar) * ldb + ah;
  const float*  gBnn = (const float*)Bv + (size_t)bn + bs;

  float4 breg[4];

  // ---- prologue: stage k0 = 0 into buffer 0 ----
  async_copy_32B(ldsA[0], gA);
  if (NT) {
    async_copy_32B(ldsB[0], gBnt);
  } else {
    const float* src = gBnn + (size_t)bk * ldb;
    #pragma unroll
    for (int q = 0; q < 4; ++q) breg[q] = *(const float4*)(src + 4 * q);
    #pragma unroll
    for (int q = 0; q < 4; ++q) {
      Bs[0][(bs + q*4 + 0) * LDSS + bk] = (__bf16)breg[q].x;
      Bs[0][(bs + q*4 + 1) * LDSS + bk] = (__bf16)breg[q].y;
      Bs[0][(bs + q*4 + 2) * LDSS + bk] = (__bf16)breg[q].z;
      Bs[0][(bs + q*4 + 3) * LDSS + bk] = (__bf16)breg[q].w;
    }
  }
  wait_async0();
  __syncthreads();

  int cur = 0;
  for (int k0 = 0; k0 < K; k0 += TILE_K) {
    const int nk  = k0 + TILE_K;
    const int nxt = cur ^ 1;
    const bool has_next = nk < K;

    // Issue next tile's movement before computing on the current one.
    if (has_next) {
      async_copy_32B(ldsA[nxt], gA + nk);
      if (NT) {
        async_copy_32B(ldsB[nxt], gBnt + nk);
      } else {
        const float* src = gBnn + (size_t)(nk + bk) * ldb;
        #pragma unroll
        for (int q = 0; q < 4; ++q) breg[q] = *(const float4*)(src + 4 * q);
      }
    }

    // Compute on buffer `cur`
    v16bf af[2], bfr[4];
    #pragma unroll
    for (int mi = 0; mi < 2; ++mi)
      af[mi] = load_frag_a(&As[cur][(wm * 32 + mi * 16) * LDSS], lane);
    #pragma unroll
    for (int ni = 0; ni < 4; ++ni)
      bfr[ni] = load_frag_b(&Bs[cur][(wn * 64 + ni * 16) * LDSS], lane);

    #pragma unroll
    for (int mi = 0; mi < 2; ++mi)
      #pragma unroll
      for (int ni = 0; ni < 4; ++ni)
        acc[mi][ni] = __builtin_amdgcn_wmma_f32_16x16x32_bf16(
            false, af[mi], false, bfr[ni], (short)0, acc[mi][ni], false, false);

    // NN: convert + store next B tile (global loads have been in flight
    // throughout the WMMA block above).
    if (!NT && has_next) {
      #pragma unroll
      for (int q = 0; q < 4; ++q) {
        Bs[nxt][(bs + q*4 + 0) * LDSS + bk] = (__bf16)breg[q].x;
        Bs[nxt][(bs + q*4 + 1) * LDSS + bk] = (__bf16)breg[q].y;
        Bs[nxt][(bs + q*4 + 2) * LDSS + bk] = (__bf16)breg[q].z;
        Bs[nxt][(bs + q*4 + 3) * LDSS + bk] = (__bf16)breg[q].w;
      }
    }

    wait_async0();
    __syncthreads();
    cur = nxt;
  }

  // Epilogue: C/D layout -> lane L: col = L%16; VGPR v: row = v + (L>=16?8:0)
  const int col0    = bn + wn * 64 + (lane & 15);
  const int rowbase = bm + wm * 32 + ((lane >> 4) << 3);
  #pragma unroll
  for (int mi = 0; mi < 2; ++mi) {
    #pragma unroll
    for (int ni = 0; ni < 4; ++ni) {
      const int col  = col0 + ni * 16;
      const int row0 = rowbase + mi * 16;
      #pragma unroll
      for (int v = 0; v < 8; ++v)
        C[(size_t)(row0 + v) * ldc + col] = acc[mi][ni][v];
    }
  }
}

// ---------------------------------------------------------------------------
// RMSNorm: out_bf16[row,:] = (x * rsqrt(mean(x^2)+eps)) * w
// ---------------------------------------------------------------------------
__global__ __launch_bounds__(256)
void rmsnorm_kernel(const float* __restrict__ x, const float* __restrict__ w,
                    __bf16* __restrict__ out, int Hd)
{
  __shared__ float red[256];
  const int row = blockIdx.x;
  const float* xr = x + (size_t)row * Hd;
  float ss = 0.f;
  for (int i = threadIdx.x; i < Hd; i += 256) { float v = xr[i]; ss += v * v; }
  red[threadIdx.x] = ss; __syncthreads();
  for (int s = 128; s > 0; s >>= 1) {
    if (threadIdx.x < s) red[threadIdx.x] += red[threadIdx.x + s];
    __syncthreads();
  }
  const float inv = rsqrtf(red[0] / (float)Hd + 1e-5f);
  for (int i = threadIdx.x; i < Hd; i += 256)
    out[(size_t)row * Hd + i] = (__bf16)(xr[i] * inv * w[i]);
}

// ---------------------------------------------------------------------------
// NeoX RoPE (rotate halves) + optional 1/sqrt(HD) scale, fp32 -> bf16
// ---------------------------------------------------------------------------
__global__ __launch_bounds__(256)
void rope_kernel(const float* __restrict__ src, __bf16* __restrict__ dst,
                 const int* __restrict__ pos, int nheads, float scale)
{
  const int s = blockIdx.x;
  const float p = (float)pos[s];
  const int npairs = nheads * (HD_ / 2);
  const float* row = src + (size_t)s * nheads * HD_;
  __bf16* orow     = dst + (size_t)s * nheads * HD_;
  for (int t = threadIdx.x; t < npairs; t += 256) {
    const int head = t >> 6, i = t & 63;
    const float inv_freq = __expf(-0.14391156f * (float)i);   // ln(1e4)/64
    float sn, cs;
    __sincosf(p * inv_freq, &sn, &cs);
    const int i1 = head * HD_ + i, i2 = i1 + HD_ / 2;
    const float x1 = row[i1], x2 = row[i2];
    orow[i1] = (__bf16)((x1 * cs - x2 * sn) * scale);
    orow[i2] = (__bf16)((x2 * cs + x1 * sn) * scale);
  }
}

// ---------------------------------------------------------------------------
// Causal row softmax: probs_bf16[i,j] = softmax(scores[i,0..i]) , 0 for j>i
// ---------------------------------------------------------------------------
__global__ __launch_bounds__(256)
void softmax_causal_kernel(const float* __restrict__ scores,
                           __bf16* __restrict__ probs, int Sd)
{
  __shared__ float red[256];
  const int i = blockIdx.x;
  const float* r = scores + (size_t)i * Sd;
  __bf16* o = probs + (size_t)i * Sd;
  const int lim = i + 1;

  float m = -3.4e38f;
  for (int j = threadIdx.x; j < lim; j += 256) m = fmaxf(m, r[j]);
  red[threadIdx.x] = m; __syncthreads();
  for (int s = 128; s > 0; s >>= 1) {
    if (threadIdx.x < s) red[threadIdx.x] = fmaxf(red[threadIdx.x], red[threadIdx.x + s]);
    __syncthreads();
  }
  m = red[0]; __syncthreads();

  float sum = 0.f;
  for (int j = threadIdx.x; j < lim; j += 256) sum += __expf(r[j] - m);
  red[threadIdx.x] = sum; __syncthreads();
  for (int s = 128; s > 0; s >>= 1) {
    if (threadIdx.x < s) red[threadIdx.x] += red[threadIdx.x + s];
    __syncthreads();
  }
  const float inv = 1.0f / red[0];

  for (int j = threadIdx.x; j < Sd; j += 256)
    o[j] = (__bf16)((j < lim) ? __expf(r[j] - m) * inv : 0.0f);
}

// ---------------------------------------------------------------------------
// Gate: logits = h @ w_gate ; softmax over E=8; top-2 renormalized -> combine
// ---------------------------------------------------------------------------
__global__ __launch_bounds__(256)
void gate_topk_kernel(const __bf16* __restrict__ h, const float* __restrict__ wg,
                      float* __restrict__ combine, int Hd)
{
  __shared__ float red[256][E_];
  const int t = blockIdx.x;
  const __bf16* hr = h + (size_t)t * Hd;
  float acc[E_] = {};
  for (int i = threadIdx.x; i < Hd; i += 256) {
    const float hv = (float)hr[i];
    const float* w = wg + (size_t)i * E_;
    #pragma unroll
    for (int e = 0; e < E_; ++e) acc[e] += hv * w[e];
  }
  #pragma unroll
  for (int e = 0; e < E_; ++e) red[threadIdx.x][e] = acc[e];
  __syncthreads();
  for (int s = 128; s > 0; s >>= 1) {
    if (threadIdx.x < s)
      for (int e = 0; e < E_; ++e) red[threadIdx.x][e] += red[threadIdx.x + s][e];
    __syncthreads();
  }
  if (threadIdx.x == 0) {
    float lg[E_]; float mx = -3.4e38f;
    for (int e = 0; e < E_; ++e) { lg[e] = red[0][e]; mx = fmaxf(mx, lg[e]); }
    float sum = 0.f;
    for (int e = 0; e < E_; ++e) { lg[e] = __expf(lg[e] - mx); sum += lg[e]; }
    for (int e = 0; e < E_; ++e) lg[e] /= sum;
    int i1 = 0;
    for (int e = 1; e < E_; ++e) if (lg[e] > lg[i1]) i1 = e;
    int i2 = -1;
    for (int e = 0; e < E_; ++e) if (e != i1 && (i2 < 0 || lg[e] > lg[i2])) i2 = e;
    const float s2 = lg[i1] + lg[i2];
    float* cr = combine + (size_t)t * E_;
    for (int e = 0; e < E_; ++e) cr[e] = 0.0f;
    cr[i1] = lg[i1] / s2;
    cr[i2] = lg[i2] / s2;
  }
}

// ---------------------------------------------------------------------------
// Elementwise helpers
// ---------------------------------------------------------------------------
__global__ void f32_to_bf16_kernel(const float* __restrict__ src,
                                   __bf16* __restrict__ dst, size_t n) {
  size_t i = (size_t)blockIdx.x * blockDim.x + threadIdx.x;
  if (i < n) dst[i] = (__bf16)src[i];
}
__global__ void add_kernel(const float* __restrict__ a, const float* __restrict__ b,
                           float* __restrict__ o, size_t n) {
  size_t i = (size_t)blockIdx.x * blockDim.x + threadIdx.x;
  if (i < n) o[i] = a[i] + b[i];
}
__global__ void zero_kernel(float* __restrict__ o, size_t n) {
  size_t i = (size_t)blockIdx.x * blockDim.x + threadIdx.x;
  if (i < n) o[i] = 0.0f;
}
__global__ void silu_mul_kernel(const float* __restrict__ a, const float* __restrict__ b,
                                __bf16* __restrict__ g, size_t n) {
  size_t i = (size_t)blockIdx.x * blockDim.x + threadIdx.x;
  if (i < n) { float v = a[i]; g[i] = (__bf16)(v / (1.0f + __expf(-v)) * b[i]); }
}
__global__ void weighted_add_kernel(float* __restrict__ out, const float* __restrict__ src,
                                    const float* __restrict__ combine, int e,
                                    int Hd, size_t n) {
  size_t i = (size_t)blockIdx.x * blockDim.x + threadIdx.x;
  if (i < n) out[i] += combine[(i / (size_t)Hd) * E_ + e] * src[i];
}

// ---------------------------------------------------------------------------
// Host orchestration
// ---------------------------------------------------------------------------
static inline void gemm_nn(const __bf16* A, int lda, const float* B, int ldb,
                           float* C, int ldc, int M, int N, int K, hipStream_t st) {
  dim3 g(N / TILE_N, M / TILE_M);
  gemm_wmma_kernel<false><<<g, 256, 0, st>>>(A, lda, (const void*)B, ldb, C, ldc, M, N, K);
}
static inline void gemm_nt(const __bf16* A, int lda, const __bf16* B, int ldb,
                           float* C, int ldc, int M, int N, int K, hipStream_t st) {
  dim3 g(N / TILE_N, M / TILE_M);
  gemm_wmma_kernel<true><<<g, 256, 0, st>>>(A, lda, (const void*)B, ldb, C, ldc, M, N, K);
}
static inline dim3 ew_grid(size_t n) { return dim3((unsigned)((n + 255) / 256)); }

extern "C" void kernel_launch(void* const* d_in, const int* in_sizes, int n_in,
                              void* d_out, int out_size, void* d_ws, size_t ws_size,
                              hipStream_t stream)
{
  (void)in_sizes; (void)n_in; (void)out_size; (void)ws_size;

  const int*   positions = (const int*)  d_in[0];
  const float* x         = (const float*)d_in[1];   // [S,H]
  const float* wq        = (const float*)d_in[2];   // [H, NH*HD]
  const float* wk        = (const float*)d_in[3];   // [H, NKV*HD]
  const float* wv        = (const float*)d_in[4];   // [H, NKV*HD]
  const float* wo        = (const float*)d_in[5];   // [NH*HD, H]
  const float* w_gate    = (const float*)d_in[6];   // [H, E]
  const float* w1        = (const float*)d_in[7];   // [E, H, F]
  const float* w2        = (const float*)d_in[8];   // [E, F, H]
  const float* w3        = (const float*)d_in[9];   // [E, H, F]
  const float* ln_in     = (const float*)d_in[10];
  const float* ln_post   = (const float*)d_in[11];

  float* moe_out   = (float*)d_out;                       // [S,H]
  float* resid_out = (float*)d_out + (size_t)T_ * H_;     // [S,H]

  // --- bump allocator over workspace ---
  char* base = (char*)d_ws;
  size_t off = 0;
  auto alloc = [&](size_t bytes) -> char* {
    char* p = base + off;
    off += (bytes + 255) & ~(size_t)255;
    return p;
  };

  __bf16* hb      = (__bf16*)alloc((size_t)T_ * H_ * 2);   // h / h2 (bf16)
  float*  combine = (float*) alloc((size_t)T_ * E_ * 4);
  const size_t pool = off;

  // ---- attention-phase pool ----
  float*  qf  = (float*) alloc((size_t)S_ * NH_  * HD_ * 4);
  float*  kf  = (float*) alloc((size_t)S_ * NKV_ * HD_ * 4);
  float*  vf  = (float*) alloc((size_t)S_ * NKV_ * HD_ * 4);
  __bf16* qb  = (__bf16*)alloc((size_t)S_ * NH_  * HD_ * 2);
  __bf16* kb  = (__bf16*)alloc((size_t)S_ * NKV_ * HD_ * 2);
  float*  sc  = (float*) alloc((size_t)S_ * S_ * 4);
  __bf16* pb  = (__bf16*)alloc((size_t)S_ * S_ * 2);
  float*  of  = (float*) alloc((size_t)S_ * NH_ * HD_ * 4);
  __bf16* ob  = (__bf16*)alloc((size_t)S_ * NH_ * HD_ * 2);
  float*  ap  = (float*) alloc((size_t)S_ * H_ * 4);

  // 1) h = rmsnorm(x, ln_in)  (bf16)
  rmsnorm_kernel<<<T_, 256, 0, stream>>>(x, ln_in, hb, H_);

  // 2) QKV projections (fp32 out)
  gemm_nn(hb, H_, wq, NH_  * HD_, qf, NH_  * HD_, S_, NH_  * HD_, H_, stream);
  gemm_nn(hb, H_, wk, NKV_ * HD_, kf, NKV_ * HD_, S_, NKV_ * HD_, H_, stream);
  gemm_nn(hb, H_, wv, NKV_ * HD_, vf, NKV_ * HD_, S_, NKV_ * HD_, H_, stream);

  // 3) RoPE; fold 1/sqrt(HD) into q; emit bf16
  rope_kernel<<<S_, 256, 0, stream>>>(qf, qb, positions, NH_,  0.08838834764831845f);
  rope_kernel<<<S_, 256, 0, stream>>>(kf, kb, positions, NKV_, 1.0f);

  // 4) Attention, one head at a time (scores stay L2-resident)
  for (int h = 0; h < NH_; ++h) {
    const int kv = h / (NH_ / NKV_);
    gemm_nt(qb + (size_t)h  * HD_, NH_  * HD_,
            kb + (size_t)kv * HD_, NKV_ * HD_,
            sc, S_, S_, S_, HD_, stream);
    softmax_causal_kernel<<<S_, 256, 0, stream>>>(sc, pb, S_);
    gemm_nn(pb, S_, vf + (size_t)kv * HD_, NKV_ * HD_,
            of + (size_t)h * HD_, NH_ * HD_, S_, HD_, S_, stream);
  }

  // 5) Output projection, residual add (residual written straight to d_out)
  f32_to_bf16_kernel<<<ew_grid((size_t)S_ * NH_ * HD_), 256, 0, stream>>>(
      of, ob, (size_t)S_ * NH_ * HD_);
  gemm_nn(ob, NH_ * HD_, wo, H_, ap, H_, S_, H_, NH_ * HD_, stream);
  add_kernel<<<ew_grid((size_t)S_ * H_), 256, 0, stream>>>(ap, x, resid_out,
                                                           (size_t)S_ * H_);

  // 6) Post-attention RMSNorm -> h2 (bf16, reuses hb)
  rmsnorm_kernel<<<T_, 256, 0, stream>>>(resid_out, ln_post, hb, H_);

  // 7) Router: softmax + top-2 renormalized combine weights
  gate_topk_kernel<<<T_, 256, 0, stream>>>(hb, w_gate, combine, H_);

  // 8) MoE (dense over experts, combine-weighted) — alias over attention pool
  off = pool;
  float*  a1 = (float*) alloc((size_t)T_ * F_ * 4);
  float*  a3 = (float*) alloc((size_t)T_ * F_ * 4);
  __bf16* gb = (__bf16*)alloc((size_t)T_ * F_ * 2);
  float*  oe = (float*) alloc((size_t)T_ * H_ * 4);

  zero_kernel<<<ew_grid((size_t)T_ * H_), 256, 0, stream>>>(moe_out, (size_t)T_ * H_);

  for (int e = 0; e < E_; ++e) {
    const float* w1e = w1 + (size_t)e * H_ * F_;
    const float* w3e = w3 + (size_t)e * H_ * F_;
    const float* w2e = w2 + (size_t)e * F_ * H_;
    gemm_nn(hb, H_, w1e, F_, a1, F_, T_, F_, H_, stream);
    gemm_nn(hb, H_, w3e, F_, a3, F_, T_, F_, H_, stream);
    silu_mul_kernel<<<ew_grid((size_t)T_ * F_), 256, 0, stream>>>(
        a1, a3, gb, (size_t)T_ * F_);
    gemm_nn(gb, F_, w2e, H_, oe, H_, T_, H_, F_, stream);
    weighted_add_kernel<<<ew_grid((size_t)T_ * H_), 256, 0, stream>>>(
        moe_out, oe, combine, e, H_, (size_t)T_ * H_);
  }
}
